// LocalAttention1D_53987738910745
// MI455X (gfx1250) — matrix operand order repaired and checked by reading, not compiled, not run on hardware
//
#include <hip/hip_runtime.h>

// ---------------------------------------------------------------------------
// LocalAttention1D for MI455X (gfx1250): bf16 WMMA everywhere.
//   D_MODEL=512, HEADS=8, HDIM=64, T=4096, B=2, WIN=16 (33-wide window)
// GEMMs stage the 64x512 weight slab in LDS via the Tensor Data Mover
// (one DMA per block, hardware row-padding), stream A rows from global with
// prefetch, and do all math on v_wmma_f32_16x16x32_bf16.
// ---------------------------------------------------------------------------

#define D_MODEL 512
#define HEADS   8
#define HDIM    64
#define TLEN    4096
#define BATCH   2
#define WINR    16
#define MTOT    (BATCH * TLEN)   // 8192 rows
#define LN_EPS  1e-5f

typedef __bf16 bf16;
typedef __attribute__((ext_vector_type(16))) __bf16 v16bf;
typedef __attribute__((ext_vector_type(8)))  __bf16 v8bf;
typedef __attribute__((ext_vector_type(8)))  float  v8f;

union V16 { v16bf v; v8bf h[2]; bf16 e[16]; };
union U8B { v8bf v; bf16 e[8]; };

// TDM availability: 5-arg builtin on ROCm (clang-22), 6-arg on upstream
// clang-23+ (therock headers). Fallback: manual vector staging.
#if defined(__has_builtin)
#  if __has_builtin(__builtin_amdgcn_tensor_load_to_lds) && \
      __has_builtin(__builtin_amdgcn_s_wait_tensorcnt)
#    define USE_TDM 1
#    if __clang_major__ >= 23
#      define TDM_6ARG 1
#    else
#      define TDM_6ARG 0
#    endif
#  endif
#endif
#ifndef USE_TDM
#  define USE_TDM 0
#endif

typedef unsigned int u32x4 __attribute__((ext_vector_type(4)));
typedef int          i32x4 __attribute__((ext_vector_type(4)));
typedef int          i32x8 __attribute__((ext_vector_type(8)));

static __device__ __forceinline__ v8f wmma_bf16(v16bf a, v16bf b, v8f c) {
  return __builtin_amdgcn_wmma_f32_16x16x32_bf16(false, a, false, b,
                                                 (short)0, c, false, false);
}

// ---------------------------------------------------------------------------
// f32 -> bf16 conversion
// ---------------------------------------------------------------------------
__global__ __launch_bounds__(256)
void cvt_f32_bf16(const float* __restrict__ in, bf16* __restrict__ out, int n) {
  int i = blockIdx.x * 256 + threadIdx.x;
  if (i < n) out[i] = (bf16)in[i];
}

// ---------------------------------------------------------------------------
// GEMM: Y = A(Mx512,bf16 row-major) * W(512x512,bf16 row-major)^T
// 256 threads = 8 waves; block tile 128(M) x 64(N); wave tile 16 x 64.
// The block's 64x512 W slab is staged in LDS once (padded rows: bank-safe).
// MODE 0: store bf16 row-major                    (Q, K)
// MODE 1: store bf16 head-transposed vT[b][n][t]  (V)
// MODE 2: store f32 row-major with residual add   (proj + residual)
// ---------------------------------------------------------------------------
template <int MODE>
__global__ __launch_bounds__(256)
void gemm_bf16_nt(const bf16* __restrict__ A, const bf16* __restrict__ W,
                  bf16* __restrict__ outb, const float* __restrict__ resid,
                  float* __restrict__ outf) {
  constexpr int LDSS = D_MODEL + 8;            // 520 halfs: row stride 4 dwords mod 64
  __shared__ bf16 sW[64 * LDSS];

  const int lane = threadIdx.x & 31;
  const int wave = threadIdx.x >> 5;
  const int l15  = lane & 15;
  const int lhi  = lane >> 4;
  const int m0   = blockIdx.x * 128 + wave * 16;
  const int n0   = blockIdx.y * 64;

#if USE_TDM
  // ---- W slab via Tensor Data Mover: one DMA, HW-inserted 16B row pad ----
  if (wave == 0) {
    const unsigned long long ga =
        (unsigned long long)(const void*)(W + (size_t)n0 * D_MODEL);
    const unsigned lds = (unsigned)(size_t)(void*)sW;   // wave-relative LDS byte addr
    // D# group0: count=1 | lds_addr | global_addr[56:0] | type=2
    u32x4 g0 = { 1u, lds,
                 (unsigned)(ga & 0xFFFFFFFFu),
                 (unsigned)((ga >> 32) & 0x01FFFFFFu) | 0x80000000u };
    // D# group1: data_size=1(2B) | pad_enable | pad_interval=7(256dw) |
    //            pad_amount=3(4dw) | tensor 512x64 | tile 512x64 | stride0=512
    i32x8 g1 = { (int)((1u << 16) | (1u << 20) | (7u << 22) | (3u << 25)),
                 (int)(512u << 16),   // tensor_dim0 = 512   (bits [79:48])
                 (int)(64u  << 16),   // tensor_dim1 = 64    (bits [111:80])
                 (int)(512u << 16),   // tile_dim0   = 512   (bits [127:112])
                 64,                  // tile_dim1   = 64    (bits [143:128])
                 512,                 // tensor_dim0_stride  (bits [207:160])
                 0, 0 };
    i32x4 z4 = { 0, 0, 0, 0 };        // groups 2/3 unused (2-D tile)
#if TDM_6ARG
    i32x8 z8 = { 0, 0, 0, 0, 0, 0, 0, 0 };
    __builtin_amdgcn_tensor_load_to_lds(g0, g1, z4, z4, z8, 0);
#else
    __builtin_amdgcn_tensor_load_to_lds(g0, g1, z4, z4, 0);
#endif
    __builtin_amdgcn_s_wait_tensorcnt(0);
  }
#else
  // ---- cooperative W slab load: 64 rows x 512 halfs (16 x 16B per thread) --
  {
    const int tid = threadIdx.x;
#pragma unroll
    for (int it = 0; it < 16; ++it) {
      const int chunk = tid + it * 256;        // 0..4095
      const int row   = chunk >> 6;
      const int col   = (chunk & 63) * 8;
      v8bf d = *(const v8bf*)(W + (size_t)(n0 + row) * D_MODEL + col);
      *(v8bf*)(&sW[row * LDSS + col]) = d;
    }
  }
#endif
  __syncthreads();

  v8f acc[4] = {{}, {}, {}, {}};

  const bf16* arow = A + (size_t)(m0 + l15) * D_MODEL;
  for (int k0 = 0; k0 < D_MODEL; k0 += 32) {
    V16 a;
    const int ka = k0 + 8 * lhi;               // A 16x32 layout: {0-7,16-23}/{8-15,24-31}
    a.h[0] = *(const v8bf*)(arow + ka);
    a.h[1] = *(const v8bf*)(arow + ka + 16);
    if (k0 + 32 < D_MODEL) __builtin_prefetch(arow + ka + 32, 0, 3);
#pragma unroll
    for (int nt = 0; nt < 4; ++nt) {
      // B(k,n) = W[n][k] from LDS: lane = n, k {0-7,16-23}/{8-15,24-31} per lane half
      const bf16* wrow = &sW[(nt * 16 + l15) * LDSS + k0 + 8 * lhi];
      V16 b;
      b.h[0] = *(const v8bf*)(wrow);
      b.h[1] = *(const v8bf*)(wrow + 16);
      acc[nt] = wmma_bf16(a.v, b.v, acc[nt]);
    }
  }

  // D 16x16 f32 layout: lane<16 -> N=lane, M=vgpr; lane>=16 -> N=lane-16, M=vgpr+8
#pragma unroll
  for (int nt = 0; nt < 4; ++nt) {
    const int n = n0 + nt * 16 + l15;
    if (MODE == 0) {
#pragma unroll
      for (int v = 0; v < 8; ++v) {
        const int m = m0 + v + 8 * lhi;
        outb[(size_t)m * D_MODEL + n] = (bf16)acc[nt][v];
      }
    } else if (MODE == 1) {
      const int b  = m0 / TLEN;
      const int tb = (m0 % TLEN) + 8 * lhi;    // 8 consecutive tokens
      U8B u;
#pragma unroll
      for (int v = 0; v < 8; ++v) u.e[v] = (bf16)acc[nt][v];
      *(v8bf*)(outb + ((size_t)(b * D_MODEL + n)) * TLEN + tb) = u.v;
    } else {
#pragma unroll
      for (int v = 0; v < 8; ++v) {
        const int m = m0 + v + 8 * lhi;
        const size_t idx = (size_t)m * D_MODEL + n;
        outf[idx] = acc[nt][v] + resid[idx];
      }
    }
  }
}

// ---------------------------------------------------------------------------
// Windowed attention, one wave per (b, h, 16-query tile).
// S^T = K_win(48x64) x Q^T(64x16)  -> softmax per-lane + shfl_xor(16)
// O   = P(16x48..pad64) x V(48..64 x 64) using head-transposed vT.
// ---------------------------------------------------------------------------
__global__ __launch_bounds__(128)
void attn_kernel(const bf16* __restrict__ qh, const bf16* __restrict__ kh,
                 const bf16* __restrict__ vT, bf16* __restrict__ attn) {
  const int lane = threadIdx.x & 31;
  const int wave = threadIdx.x >> 5;
  const int l15  = lane & 15;
  const int lhi  = lane >> 4;

  const int g    = blockIdx.x * 4 + wave;      // 0..4095
  const int tile = g & 255;                    // T/16 tiles
  const int bh   = g >> 8;
  const int h    = bh & 7;
  const int b    = bh >> 3;
  const int t0   = tile * 16;

  // ---- B fragments of Q^T: lane = query column, contiguous dims ----
  const bf16* qrow = qh + ((size_t)(b * TLEN + t0 + l15)) * D_MODEL + h * HDIM;
  V16 bq[2];
  bq[0].v = *(const v16bf*)(qrow + 16 * lhi);
  bq[1].v = *(const v16bf*)(qrow + 32 + 16 * lhi);

  // ---- S^T tiles: A = K window rows ----
  v8f s[3] = {{}, {}, {}};
#pragma unroll
  for (int j = 0; j < 3; ++j) {
    int tok = t0 - WINR + j * 16 + l15;
    tok = min(max(tok, 0), TLEN - 1);          // clamp like reference idx
    const bf16* krow = kh + ((size_t)(b * TLEN + tok)) * D_MODEL + h * HDIM;
#pragma unroll
    for (int c = 0; c < 2; ++c) {
      V16 ak;
      const int ka = 32 * c + 16 * lhi;
      ak.h[0] = *(const v8bf*)(krow + ka);
      ak.h[1] = *(const v8bf*)(krow + ka + 8);
      s[j] = wmma_bf16(ak.v, bq[c].v, s[j]);
    }
  }

  // S^T frag j: this lane owns query m=l15 (t = t0+l15), window w = 16j+v+8*lhi
  const float scale = 0.125f;                  // 64^-0.5
  float p[3][8];
  float mx = -__builtin_inff();
#pragma unroll
  for (int j = 0; j < 3; ++j)
#pragma unroll
    for (int v = 0; v < 8; ++v) {
      const int w   = 16 * j + v + 8 * lhi;
      const int pos = t0 - WINR + w;           // absolute key token
      const bool valid = (w >= l15) && (w <= l15 + 2 * WINR) &&
                         (pos >= 0) && (pos < TLEN);
      float sv = s[j][v] * scale;
      if (!valid) sv = -__builtin_inff();
      p[j][v] = sv;
      mx = fmaxf(mx, sv);
    }
  mx = fmaxf(mx, __shfl_xor(mx, 16, 32));      // combine the two half-rows

  float sum = 0.f;
#pragma unroll
  for (int j = 0; j < 3; ++j)
#pragma unroll
    for (int v = 0; v < 8; ++v) {
      p[j][v] = __expf(p[j][v] - mx);
      sum += p[j][v];
    }
  sum += __shfl_xor(sum, 16, 32);
  const float inv = 1.0f / sum;

  // ---- P^T frags -> A frags of P (pure in-lane repack) ----
  V16 aP0, aP1;
#pragma unroll
  for (int v = 0; v < 8; ++v) {
    aP0.e[v]     = (bf16)(p[0][v] * inv);      // window k 0-7 / 8-15
    aP0.e[v + 8] = (bf16)(p[1][v] * inv);      // window k 16-23 / 24-31
    aP1.e[v]     = (bf16)(p[2][v] * inv);      // window k 32-39 / 40-47
    aP1.e[v + 8] = (bf16)0.0f;                 // pad 48-63
  }

  // ---- O = P x V via head-transposed vT (contiguous token runs) ----
  const bf16* vbase = vT + ((size_t)(b * D_MODEL + h * HDIM)) * TLEN;
  v8f o[4] = {{}, {}, {}, {}};
#pragma unroll
  for (int nt = 0; nt < 4; ++nt) {
    const int n = nt * 16 + l15;               // head dim, fixed per lane
#pragma unroll
    for (int c = 0; c < 2; ++c) {
      int tb = t0 - WINR + 32 * c + 8 * lhi;   // 8-token segments at tb, tb+16
      const int tb0 = min(max(tb, 0), TLEN - 8);        // clamped seg feeds only P==0 lanes
      const int tb1 = min(max(tb + 16, 0), TLEN - 8);
      V16 bv;
      bv.h[0] = *(const v8bf*)(vbase + (size_t)n * TLEN + tb0);
      bv.h[1] = *(const v8bf*)(vbase + (size_t)n * TLEN + tb1);
      o[nt] = wmma_bf16(c == 0 ? aP0.v : aP1.v, bv.v, o[nt]);
    }
  }

  // store O row-major bf16: lane owns column h*64+n, rows t0+v+8*lhi
#pragma unroll
  for (int nt = 0; nt < 4; ++nt) {
    const int n = h * HDIM + nt * 16 + l15;
#pragma unroll
    for (int v = 0; v < 8; ++v) {
      const int t = t0 + v + 8 * lhi;
      attn[((size_t)(b * TLEN + t)) * D_MODEL + n] = (bf16)o[nt][v];
    }
  }
}

// ---------------------------------------------------------------------------
// In-place LayerNorm over 512 f32 per row; one 256-thread block per row.
// ---------------------------------------------------------------------------
__global__ __launch_bounds__(256)
void ln_inplace(float* __restrict__ y, const float* __restrict__ gamma,
                const float* __restrict__ beta) {
  __shared__ float sred[8];
  const int tid = threadIdx.x;
  float* row = y + (size_t)blockIdx.x * D_MODEL;
  const float a0 = row[tid];
  const float a1 = row[tid + 256];

  float s = a0 + a1;
#pragma unroll
  for (int o = 16; o > 0; o >>= 1) s += __shfl_xor(s, o, 32);
  if ((tid & 31) == 0) sred[tid >> 5] = s;
  __syncthreads();
  float tot = 0.f;
#pragma unroll
  for (int i = 0; i < 8; ++i) tot += sred[i];
  const float mu = tot * (1.0f / D_MODEL);
  __syncthreads();

  const float d0 = a0 - mu, d1 = a1 - mu;
  float vs = d0 * d0 + d1 * d1;
#pragma unroll
  for (int o = 16; o > 0; o >>= 1) vs += __shfl_xor(vs, o, 32);
  if ((tid & 31) == 0) sred[tid >> 5] = vs;
  __syncthreads();
  float vtot = 0.f;
#pragma unroll
  for (int i = 0; i < 8; ++i) vtot += sred[i];
  const float rs = rsqrtf(vtot * (1.0f / D_MODEL) + LN_EPS);

  row[tid]       = gamma[tid]       * d0 * rs + beta[tid];
  row[tid + 256] = gamma[tid + 256] * d1 * rs + beta[tid + 256];
}

// ---------------------------------------------------------------------------
extern "C" void kernel_launch(void* const* d_in, const int* in_sizes, int n_in,
                              void* d_out, int out_size, void* d_ws, size_t ws_size,
                              hipStream_t stream) {
  const float* x     = (const float*)d_in[0];
  const float* Wq    = (const float*)d_in[1];
  const float* Wk    = (const float*)d_in[2];
  const float* Wv    = (const float*)d_in[3];
  const float* Wp    = (const float*)d_in[4];
  const float* gamma = (const float*)d_in[5];
  const float* beta  = (const float*)d_in[6];
  float* out = (float*)d_out;

  char* ws = (char*)d_ws;
  size_t off = 0;
  auto alloc = [&](size_t bytes) { char* p = ws + off; off += (bytes + 255) & ~(size_t)255; return p; };

  bf16* xh   = (bf16*)alloc((size_t)MTOT * D_MODEL * 2);
  bf16* whq  = (bf16*)alloc((size_t)D_MODEL * D_MODEL * 2);
  bf16* whk  = (bf16*)alloc((size_t)D_MODEL * D_MODEL * 2);
  bf16* whv  = (bf16*)alloc((size_t)D_MODEL * D_MODEL * 2);
  bf16* whp  = (bf16*)alloc((size_t)D_MODEL * D_MODEL * 2);
  bf16* qh   = (bf16*)alloc((size_t)MTOT * D_MODEL * 2);
  bf16* kh   = (bf16*)alloc((size_t)MTOT * D_MODEL * 2);
  bf16* vT   = (bf16*)alloc((size_t)MTOT * D_MODEL * 2);   // [b][h*64+dim][T]
  bf16* atn  = (bf16*)alloc((size_t)MTOT * D_MODEL * 2);

  const int nx = MTOT * D_MODEL;       // 4,194,304
  const int nw = D_MODEL * D_MODEL;    // 262,144
  cvt_f32_bf16<<<(nx + 255) / 256, 256, 0, stream>>>(x,  xh,  nx);
  cvt_f32_bf16<<<(nw + 255) / 256, 256, 0, stream>>>(Wq, whq, nw);
  cvt_f32_bf16<<<(nw + 255) / 256, 256, 0, stream>>>(Wk, whk, nw);
  cvt_f32_bf16<<<(nw + 255) / 256, 256, 0, stream>>>(Wv, whv, nw);
  cvt_f32_bf16<<<(nw + 255) / 256, 256, 0, stream>>>(Wp, whp, nw);

  dim3 ggrid(MTOT / 128, D_MODEL / 64);   // (64, 8)
  gemm_bf16_nt<0><<<ggrid, 256, 0, stream>>>(xh, whq, qh, nullptr, nullptr);
  gemm_bf16_nt<0><<<ggrid, 256, 0, stream>>>(xh, whk, kh, nullptr, nullptr);
  gemm_bf16_nt<1><<<ggrid, 256, 0, stream>>>(xh, whv, vT, nullptr, nullptr);

  attn_kernel<<<(BATCH * HEADS * (TLEN / 16)) / 4, 128, 0, stream>>>(qh, kh, vT, atn);

  gemm_bf16_nt<2><<<ggrid, 256, 0, stream>>>(atn, whp, nullptr, x, out);

  ln_inplace<<<MTOT, 256, 0, stream>>>(out, gamma, beta);
}
